// SelfAttention_60919816126899
// MI455X (gfx1250) — matrix operand order repaired and checked by reading, not compiled, not run on hardware
//
#include <hip/hip_runtime.h>
#include <hip/hip_bf16.h>

// ---------------------------------------------------------------------------
// Self-attention forward for MI455X (gfx1250, wave32, WMMA bf16).
// B=4, S=2048, E=2048, H=16, D=128.
// ---------------------------------------------------------------------------

typedef __attribute__((ext_vector_type(16))) __bf16 v16bf;
typedef __attribute__((ext_vector_type(8)))  float  v8f;
typedef int b128_t __attribute__((vector_size(16)));   // matches builtin param type

union Frag {          // one WMMA A/B operand: 16 bf16 = 32 bytes
    uint4 u4[2];
    v16bf v;
};

#if __has_builtin(__builtin_amdgcn_global_load_async_to_lds_b128) && \
    __has_builtin(__builtin_amdgcn_s_wait_asynccnt)
#define USE_ASYNC 1
#else
#define USE_ASYNC 0
#endif

__device__ __forceinline__ unsigned short f2bf(float f) {
    unsigned u = __float_as_uint(f);
    unsigned r = u + 0x7FFFu + ((u >> 16) & 1u);   // round-to-nearest-even
    return (unsigned short)(r >> 16);
}
__device__ __forceinline__ unsigned int pack2(float a, float b) {
#if __has_builtin(__builtin_amdgcn_cvt_pk_bf16_f32)
    auto pk = __builtin_amdgcn_cvt_pk_bf16_f32(a, b);   // v_cvt_pk_bf16_f32
    return __builtin_bit_cast(unsigned int, pk);
#else
    return (unsigned int)f2bf(a) | ((unsigned int)f2bf(b) << 16);
#endif
}
__device__ __forceinline__ v8f wmma_bf16(v16bf a, v16bf b, v8f c) {
    return __builtin_amdgcn_wmma_f32_16x16x32_bf16(false, a, false, b,
                                                   (short)0, c, false, false);
}

// 16-byte global -> LDS copy; async (ASYNCcnt) when available.
__device__ __forceinline__ void async_cp16(unsigned short* lds, const unsigned short* g) {
#if USE_ASYNC
    __builtin_amdgcn_global_load_async_to_lds_b128(
        (__attribute__((address_space(1))) b128_t*)g,
        (__attribute__((address_space(3))) b128_t*)lds, 0, 0);
#else
    *(uint4*)lds = *(const uint4*)g;
#endif
}

// ---------------------------------------------------------------------------
// Kernel 1: QKV projection.  O(bf16) = X(f32)[8192,2048] * W(f32)[2048,2048]
// Output layout: transposeV==0 -> [b][h][s][d], transposeV==1 -> [b][h][d][s]
// Grid (32, 64), block 256 (8 waves). Block tile 128(M) x 64(N), k-step 64.
// ---------------------------------------------------------------------------
__global__ __launch_bounds__(256)
void proj_kernel(const float* __restrict__ X, const float* __restrict__ W,
                 unsigned short* __restrict__ O, int transposeV)
{
    __shared__ __align__(16) unsigned short lA[128 * 64];   // [row][k]  16KB
    __shared__ __align__(16) unsigned short lB[64 * 64];    // [n][k]     8KB

    const int t      = threadIdx.x;
    const int w      = t >> 5;
    const int lane   = t & 31;
    const int lr     = lane & 15;
    const int hi     = lane >> 4;
    const int waveM  = w & 3;       // 4 x 32 = 128 rows
    const int waveN  = w >> 2;      // 2 x 32 = 64 cols
    const int blockN = blockIdx.x * 64;
    const int blockM = blockIdx.y * 128;

    v8f acc[2][2];
    #pragma unroll
    for (int i = 0; i < 2; ++i)
        #pragma unroll
        for (int j = 0; j < 2; ++j)
            #pragma unroll
            for (int v = 0; v < 8; ++v) acc[i][j][v] = 0.0f;

    for (int k0 = 0; k0 < 2048; k0 += 64) {
        // ---- stage A tile 128x64 (fp32 -> bf16) ----
        {
            int row  = t >> 1;
            int half = t & 1;
            const float4* src =
                (const float4*)(X + (size_t)(blockM + row) * 2048 + k0 + half * 32);
            uint4* dst = (uint4*)&lA[row * 64 + half * 32];
            #pragma unroll
            for (int q = 0; q < 4; ++q) {
                float4 f0 = src[2 * q], f1 = src[2 * q + 1];
                dst[q] = make_uint4(pack2(f0.x, f0.y), pack2(f0.z, f0.w),
                                    pack2(f1.x, f1.y), pack2(f1.z, f1.w));
            }
        }
        // ---- stage B tile as [n][k] (transpose of W[k][n], fp32 -> bf16) ----
        {
            int k  = t >> 2;             // 0..63
            int nq = (t & 3) * 16;       // 0,16,32,48
            const float* src = W + (size_t)(k0 + k) * 2048 + blockN + nq;
            #pragma unroll
            for (int j = 0; j < 16; ++j) lB[(nq + j) * 64 + k] = f2bf(src[j]);
        }
        __syncthreads();

        #pragma unroll
        for (int kc = 0; kc < 64; kc += 32) {
            Frag af[2], bfr[2];
            #pragma unroll
            for (int sm = 0; sm < 2; ++sm) {
                const uint4* p =
                    (const uint4*)&lA[(waveM * 32 + sm * 16 + lr) * 64 + kc + hi * 16];
                af[sm].u4[0] = p[0];          // K = 16*hi + 0..7
                af[sm].u4[1] = p[1];          // K = 16*hi + 8..15
            }
            #pragma unroll
            for (int sn = 0; sn < 2; ++sn) {
                const uint4* p =
                    (const uint4*)&lB[(waveN * 32 + sn * 16 + lr) * 64 + kc + hi * 16];
                bfr[sn].u4[0] = p[0];         // K = 16*hi + 0..15 (contiguous)
                bfr[sn].u4[1] = p[1];
            }
            #pragma unroll
            for (int sm = 0; sm < 2; ++sm)
                #pragma unroll
                for (int sn = 0; sn < 2; ++sn)
                    acc[sm][sn] = wmma_bf16(af[sm].v, bfr[sn].v, acc[sm][sn]);
        }
        __syncthreads();
    }

    // ---- epilogue: f32 -> bf16, head-split layout ----
    #pragma unroll
    for (int sm = 0; sm < 2; ++sm)
        #pragma unroll
        for (int sn = 0; sn < 2; ++sn)
            #pragma unroll
            for (int v = 0; v < 8; ++v) {
                int m = blockM + waveM * 32 + sm * 16 + v + hi * 8;
                int n = blockN + waveN * 32 + sn * 16 + lr;
                int bb = m >> 11, s = m & 2047, hh = n >> 7, d = n & 127;
                size_t addr = transposeV
                    ? ((size_t)((bb * 16 + hh) * 128 + d) * 2048 + s)
                    : ((size_t)((bb * 16 + hh) * 2048 + s) * 128 + d);
                O[addr] = f2bf(acc[sm][sn][v]);
            }
}

// ---------------------------------------------------------------------------
// Kernel 2: flash attention.  Grid (S/64=32, H=16, B=4), block 128 (4 waves).
// Each wave: 16 query rows, O[16][128] f32 accum, online softmax over S keys.
// K/V^T tiles (32 keys) double-buffered in LDS via async global->LDS loads.
// ---------------------------------------------------------------------------
__device__ __forceinline__ void attn_issue_tile(
    const unsigned short* __restrict__ Kg, const unsigned short* __restrict__ VT,
    size_t base, int kb, int tid, unsigned short* lKbuf, unsigned short* lVbuf)
{
    // 8 async b128 ops per thread per tile
    int i = tid >> 2, off = (tid & 3) * 32;
    const unsigned short* gk = Kg + base + (size_t)(kb + i) * 128 + off;
    unsigned short* dk = &lKbuf[i * 128 + off];
    async_cp16(dk,      gk);
    async_cp16(dk + 8,  gk + 8);
    async_cp16(dk + 16, gk + 16);
    async_cp16(dk + 24, gk + 24);
    const unsigned short* gv = VT + base + (size_t)tid * 2048 + kb;
    unsigned short* dv = &lVbuf[tid * 32];
    async_cp16(dv,      gv);
    async_cp16(dv + 8,  gv + 8);
    async_cp16(dv + 16, gv + 16);
    async_cp16(dv + 24, gv + 24);
}

__global__ __launch_bounds__(128)
void attn_kernel(const unsigned short* __restrict__ Q,
                 const unsigned short* __restrict__ Kg,
                 const unsigned short* __restrict__ VT,
                 const int* __restrict__ mask,
                 unsigned short* __restrict__ O)
{
    __shared__ __align__(16) unsigned short lK[2][32 * 128];  // [key][d]   2x8KB
    __shared__ __align__(16) unsigned short lV[2][128 * 32];  // [d][key]   2x8KB
    __shared__ __align__(16) unsigned short lP[4 * 16 * 32];  // per-wave P  4KB

    const int tid  = threadIdx.x;
    const int w    = tid >> 5;
    const int lane = tid & 31;
    const int lr   = lane & 15;
    const int hi   = lane >> 4;
    const int b    = blockIdx.z, h = blockIdx.y;
    const int q0   = blockIdx.x * 64 + w * 16;
    const size_t base = (size_t)(b * 16 + h) * 2048 * 128;
    const float scale = 0.088388347648318447f;  // 1/sqrt(128)

    // prefetch first two key tiles
    attn_issue_tile(Kg, VT, base, 0,  tid, lK[0], lV[0]);
    attn_issue_tile(Kg, VT, base, 32, tid, lK[1], lV[1]);

    // Q fragments (A-layout), kept in registers for all key blocks
    Frag qf[4];
    #pragma unroll
    for (int ks = 0; ks < 4; ++ks) {
        const uint4* p = (const uint4*)(Q + base + (size_t)(q0 + lr) * 128 + ks * 32 + hi * 16);
        qf[ks].u4[0] = p[0];
        qf[ks].u4[1] = p[1];
    }

    float mrow[8], lrow[8];
    v8f acc[8];
    #pragma unroll
    for (int v = 0; v < 8; ++v) { mrow[v] = -1e30f; lrow[v] = 0.0f; }
    #pragma unroll
    for (int dt = 0; dt < 8; ++dt)
        #pragma unroll
        for (int v = 0; v < 8; ++v) acc[dt][v] = 0.0f;

    int cur = 0;
    for (int kb = 0; kb < 2048; kb += 32) {
#if USE_ASYNC
        // 8 newer ops belong to the next tile; oldest 8 (current tile) must land.
        if (kb + 32 < 2048) __builtin_amdgcn_s_wait_asynccnt(8);
        else                __builtin_amdgcn_s_wait_asynccnt(0);
#endif
        __syncthreads();           // all waves' current tile visible

        const unsigned short* lKc = lK[cur];
        const unsigned short* lVc = lV[cur];

        // ---- S = Q * K^T for 32 keys (two 16-wide n-tiles) ----
        v8f c0, c1;
        #pragma unroll
        for (int v = 0; v < 8; ++v) { c0[v] = 0.0f; c1[v] = 0.0f; }
        #pragma unroll
        for (int ks = 0; ks < 4; ++ks) {
            Frag b0, b1;
            const uint4* p0 = (const uint4*)&lKc[lr * 128 + ks * 32 + hi * 16];
            b0.u4[0] = p0[0]; b0.u4[1] = p0[1];
            const uint4* p1 = (const uint4*)&lKc[(16 + lr) * 128 + ks * 32 + hi * 16];
            b1.u4[0] = p1[0]; b1.u4[1] = p1[1];
            c0 = wmma_bf16(qf[ks].v, b0.v, c0);
            c1 = wmma_bf16(qf[ks].v, b1.v, c1);
        }

        // ---- mask + online softmax (rows v+8*hi live in one 16-lane half) ----
        const size_t mbase = (size_t)q0 * 2048 + kb;
        #pragma unroll
        for (int v = 0; v < 8; ++v) {
            int r  = v + hi * 8;
            int m0 = mask[mbase + (size_t)r * 2048 + lr];
            int m1 = mask[mbase + (size_t)r * 2048 + 16 + lr];
            float e0 = (m0 != 0) ? c0[v] * scale : -1e20f;
            float e1 = (m1 != 0) ? c1[v] * scale : -1e20f;
            float mx = fmaxf(e0, e1);
            mx = fmaxf(mx, __shfl_xor(mx, 1, 32));
            mx = fmaxf(mx, __shfl_xor(mx, 2, 32));
            mx = fmaxf(mx, __shfl_xor(mx, 4, 32));
            mx = fmaxf(mx, __shfl_xor(mx, 8, 32));
            float nm = fmaxf(mrow[v], mx);
            float cf = __expf(mrow[v] - nm);
            mrow[v]  = nm;
            float p0v = __expf(e0 - nm);
            float p1v = __expf(e1 - nm);
            float sm = p0v + p1v;
            sm += __shfl_xor(sm, 1, 32);
            sm += __shfl_xor(sm, 2, 32);
            sm += __shfl_xor(sm, 4, 32);
            sm += __shfl_xor(sm, 8, 32);
            lrow[v] = lrow[v] * cf + sm;
            #pragma unroll
            for (int dt = 0; dt < 8; ++dt) acc[dt][v] *= cf;
            lP[w * 512 + r * 32 + lr]      = f2bf(p0v);   // C-layout -> LDS bounce
            lP[w * 512 + r * 32 + 16 + lr] = f2bf(p1v);
        }

        // ---- re-read P as A-fragment, O += P * V ----
        Frag pf;
        {
            const uint4* p = (const uint4*)&lP[w * 512 + lr * 32 + hi * 16];
            pf.u4[0] = p[0];
            pf.u4[1] = p[1];
        }
        #pragma unroll
        for (int dt = 0; dt < 8; ++dt) {
            Frag bv;
            const uint4* p = (const uint4*)&lVc[(dt * 16 + lr) * 32 + hi * 16];
            bv.u4[0] = p[0]; bv.u4[1] = p[1];
            acc[dt] = wmma_bf16(pf.v, bv.v, acc[dt]);
        }

        __syncthreads();           // everyone done reading buffer `cur`
        if (kb + 64 < 2048)        // prefetch tile kb+64 into the freed buffer
            attn_issue_tile(Kg, VT, base, kb + 64, tid, lK[cur], lV[cur]);
        cur ^= 1;
    }

    // ---- epilogue: O /= l, write bf16 to [b][s][e] ----
    #pragma unroll
    for (int v = 0; v < 8; ++v) {
        int r = v + hi * 8;
        float inv = 1.0f / lrow[v];
        size_t orow = (size_t)(b * 2048 + q0 + r) * 2048 + h * 128;
        #pragma unroll
        for (int dt = 0; dt < 8; ++dt)
            O[orow + dt * 16 + lr] = f2bf(acc[dt][v] * inv);
    }
}

// ---------------------------------------------------------------------------
// Kernel 3: output projection.  out(f32) = A(bf16)[8192,2048] * Wo(f32) + bo
// ---------------------------------------------------------------------------
__global__ __launch_bounds__(256)
void oproj_kernel(const unsigned short* __restrict__ A, const float* __restrict__ W,
                  const float* __restrict__ bias, float* __restrict__ out)
{
    __shared__ __align__(16) unsigned short lA[128 * 64];
    __shared__ __align__(16) unsigned short lB[64 * 64];

    const int t      = threadIdx.x;
    const int w      = t >> 5;
    const int lane   = t & 31;
    const int lr     = lane & 15;
    const int hi     = lane >> 4;
    const int waveM  = w & 3;
    const int waveN  = w >> 2;
    const int blockN = blockIdx.x * 64;
    const int blockM = blockIdx.y * 128;

    v8f acc[2][2];
    #pragma unroll
    for (int i = 0; i < 2; ++i)
        #pragma unroll
        for (int j = 0; j < 2; ++j)
            #pragma unroll
            for (int v = 0; v < 8; ++v) acc[i][j][v] = 0.0f;

    for (int k0 = 0; k0 < 2048; k0 += 64) {
        {   // A tile already bf16: plain 64B/thread copy
            int row  = t >> 1;
            int half = t & 1;
            const uint4* src =
                (const uint4*)(A + (size_t)(blockM + row) * 2048 + k0 + half * 32);
            uint4* dst = (uint4*)&lA[row * 64 + half * 32];
            dst[0] = src[0]; dst[1] = src[1]; dst[2] = src[2]; dst[3] = src[3];
        }
        {   // Wo fp32 -> bf16, transpose into [n][k]
            int k  = t >> 2;
            int nq = (t & 3) * 16;
            const float* src = W + (size_t)(k0 + k) * 2048 + blockN + nq;
            #pragma unroll
            for (int j = 0; j < 16; ++j) lB[(nq + j) * 64 + k] = f2bf(src[j]);
        }
        __syncthreads();

        #pragma unroll
        for (int kc = 0; kc < 64; kc += 32) {
            Frag af[2], bfr[2];
            #pragma unroll
            for (int sm = 0; sm < 2; ++sm) {
                const uint4* p =
                    (const uint4*)&lA[(waveM * 32 + sm * 16 + lr) * 64 + kc + hi * 16];
                af[sm].u4[0] = p[0];
                af[sm].u4[1] = p[1];
            }
            #pragma unroll
            for (int sn = 0; sn < 2; ++sn) {
                const uint4* p =
                    (const uint4*)&lB[(waveN * 32 + sn * 16 + lr) * 64 + kc + hi * 16];
                bfr[sn].u4[0] = p[0];
                bfr[sn].u4[1] = p[1];
            }
            #pragma unroll
            for (int sm = 0; sm < 2; ++sm)
                #pragma unroll
                for (int sn = 0; sn < 2; ++sn)
                    acc[sm][sn] = wmma_bf16(af[sm].v, bfr[sn].v, acc[sm][sn]);
        }
        __syncthreads();
    }

    #pragma unroll
    for (int sm = 0; sm < 2; ++sm)
        #pragma unroll
        for (int sn = 0; sn < 2; ++sn)
            #pragma unroll
            for (int v = 0; v < 8; ++v) {
                int m = blockM + waveM * 32 + sm * 16 + v + hi * 8;
                int n = blockN + waveN * 32 + sn * 16 + lr;
                out[(size_t)m * 2048 + n] = acc[sm][sn][v] + bias[n];
            }
}

// ---------------------------------------------------------------------------
extern "C" void kernel_launch(void* const* d_in, const int* in_sizes, int n_in,
                              void* d_out, int out_size, void* d_ws, size_t ws_size,
                              hipStream_t stream)
{
    (void)in_sizes; (void)n_in; (void)out_size; (void)ws_size;

    const float* values = (const float*)d_in[0];
    const float* keys   = (const float*)d_in[1];
    const float* query  = (const float*)d_in[2];
    const int*   mask   = (const int*)d_in[3];
    const float* Wv     = (const float*)d_in[4];
    const float* Wk     = (const float*)d_in[5];
    const float* Wq     = (const float*)d_in[6];
    const float* Wo     = (const float*)d_in[7];
    const float* bo     = (const float*)d_in[8];
    float* out          = (float*)d_out;

    const size_t NELEM = (size_t)4 * 16 * 2048 * 128;   // 16,777,216 per buffer
    unsigned short* Qb = (unsigned short*)d_ws;         // [b][h][s][d]
    unsigned short* Kb = Qb + NELEM;                    // [b][h][s][d]
    unsigned short* Vt = Kb + NELEM;                    // [b][h][d][s]
    unsigned short* AO = Vt + NELEM;                    // [b][s][e]

    dim3 gemmGrid(2048 / 64, 8192 / 128);               // (32, 64)
    proj_kernel<<<gemmGrid, 256, 0, stream>>>(query,  Wq, Qb, 0);
    proj_kernel<<<gemmGrid, 256, 0, stream>>>(keys,   Wk, Kb, 0);
    proj_kernel<<<gemmGrid, 256, 0, stream>>>(values, Wv, Vt, 1);

    attn_kernel<<<dim3(32, 16, 4), 128, 0, stream>>>(Qb, Kb, Vt, mask, AO);

    oproj_kernel<<<gemmGrid, 256, 0, stream>>>(AO, Wo, bo, out);
}
